// LSTMModel_3246995275913
// MI455X (gfx1250) — matrix operand order repaired and checked by reading, not compiled, not run on hardware
//
#include <hip/hip_runtime.h>

// ---------------------------------------------------------------------------
// LSTM for MI455X (gfx1250): bf16 WMMA GEMMs with f32 accumulation.
//   K0: convert weights (W_x2h / W_h2h / W_fc) to WMMA-B bf16 fragment layout
//   K1: Xproj = x @ W_x2h.T + (b_x2h + b_h2h), stored in C-fragment layout
//       [bg][t][n][lane][r] so the recurrence loads accumulators with b128s
//   K2: persistent per-batch-group recurrence (4 WGs, LDS h, 2 barriers/step)
//   K3: out = h_T @ W_fc.T + b_fc
// ---------------------------------------------------------------------------

typedef __attribute__((ext_vector_type(16))) __bf16       v16bf;
typedef __attribute__((ext_vector_type(8)))  float        v8f;
typedef __attribute__((ext_vector_type(4)))  float        v4f;
typedef __attribute__((ext_vector_type(4)))  unsigned int v4u;

union FragB { v16bf v; unsigned short u[16]; v4u q[2]; };

static constexpr int Bsz = 64, Tn = 512, Dn = 256, Hn = 256, G4 = 1024, On = 128;

__device__ __forceinline__ unsigned short f2bf(float f) {
  unsigned int u = __builtin_bit_cast(unsigned int, f);
  u += 0x7fffu + ((u >> 16) & 1u);   // round-to-nearest-even
  return (unsigned short)(u >> 16);
}
__device__ __forceinline__ float sigf(float x) { return 1.f / (1.f + __expf(-x)); }
__device__ __forceinline__ float tanhfast(float x) {
  float e = __expf(-2.f * x);
  return (1.f - e) / (1.f + e);
}

// ISA 7.12.2: 16-bit A-matrix 16x32 layout. For element (M,K):
//   K = 2*(v%4) + p + 8*half + 16*(v/4),  lane = half*16 + M,  e = 2v+p
__device__ __forceinline__ void amap(int M, int K, int& lane, int& e) {
  int hi = K >> 4, Km = K & 15, half = Km >> 3, rr = Km & 7;
  int v = (rr >> 1) + 4 * hi, p = rr & 1;
  lane = half * 16 + M;
  e = 2 * v + p;
}

__device__ __forceinline__ v8f wmma_bf16(const FragB& a, const FragB& b, v8f c) {
  return __builtin_amdgcn_wmma_f32_16x16x32_bf16(false, a.v, false, b.v,
                                                 (short)0, c, false, false);
}

__device__ __forceinline__ v8f join8(v4f lo, v4f hi) {
  return __builtin_shufflevector(lo, hi, 0, 1, 2, 3, 4, 5, 6, 7);
}

// ---------------------------------------------------------------------------
// K0: W [16*ntiles, 256] f32 row-major -> WMMA-B bf16 fragments
//     dst[((n*8 + kk)*32 + lane)*16 + e], B layout: N = lane&15,
//     K = (e&7) + 8*(lane>>4) + 16*(e>>3)   (within the 32-wide K block kk)
// ---------------------------------------------------------------------------
__global__ __launch_bounds__(256) void wfrag_convert(
    const float* __restrict__ W, unsigned short* __restrict__ dst, int ntiles) {
  int idx = blockIdx.x * 256 + threadIdx.x;
  if (idx >= ntiles * 4096) return;
  int e = idx & 15, lane = (idx >> 4) & 31, kk = (idx >> 9) & 7, n = idx >> 12;
  int N = lane & 15;
  int K = (e & 7) + 8 * (lane >> 4) + 16 * (e >> 3);
  dst[idx] = f2bf(W[(16 * n + N) * 256 + 32 * kk + K]);
}

// ---------------------------------------------------------------------------
// K1: Xproj = x @ W_x2h.T + b_x2h + b_h2h, written in C-fragment layout:
//     Xf[(((bg*Tn + t)*64 + n)*32 + lane)*8 + r]
// grid (2048, 8): blockIdx.x = bg*512 + t (tile rows = batch group rows at
// fixed t), blockIdx.y = col-tile block; block 256 (8 waves, 1 tile each).
// ---------------------------------------------------------------------------
__global__ __launch_bounds__(256) void xproj_kernel(
    const float* __restrict__ x, const unsigned short* __restrict__ wxb,
    const float* __restrict__ b1, const float* __restrict__ b2,
    float* __restrict__ Xf) {
  __shared__ __align__(16) unsigned short a_swz[8][32][16];  // 8 KB
  int bg = blockIdx.x >> 9, t = blockIdx.x & 511, nb = blockIdx.y;
  int tid = threadIdx.x, wave = tid >> 5, lane = tid & 31;

  // Stage 16x256 f32 x-tile (rows = batch group, fixed t) as bf16 A-fragments.
  {
    int M = tid >> 4, c0 = (tid & 15) * 16;
    const float* xrow = x + ((size_t)(bg * 16 + M) * Tn + t) * Dn + c0;
#pragma unroll
    for (int j = 0; j < 16; ++j) {
      int col = c0 + j;
      int l2, e;
      amap(M, col & 31, l2, e);
      a_swz[col >> 5][l2][e] = f2bf(xrow[j]);
    }
  }
  __syncthreads();

  int n = nb * 8 + wave;
  int Nl = lane & 15;
  float bsum = b1[16 * n + Nl] + b2[16 * n + Nl];
  v8f acc;
#pragma unroll
  for (int r = 0; r < 8; ++r) acc[r] = bsum;

#pragma unroll
  for (int kk = 0; kk < 8; ++kk) {
    FragB a, b;
    a.q[0] = *(const v4u*)&a_swz[kk][lane][0];
    a.q[1] = *(const v4u*)&a_swz[kk][lane][8];
    const v4u* wp = (const v4u*)(wxb + (((n * 8 + kk) * 32 + lane) << 4));
    b.q[0] = wp[0];
    b.q[1] = wp[1];
    acc = wmma_bf16(a, b, acc);
  }

  float* dst = Xf + ((((size_t)bg * Tn + t) * 64 + n) * 32 + lane) * 8;
  *(v4f*)dst       = __builtin_shufflevector(acc, acc, 0, 1, 2, 3);
  *((v4f*)dst + 1) = __builtin_shufflevector(acc, acc, 4, 5, 6, 7);
}

// ---------------------------------------------------------------------------
// K2: persistent recurrence. grid 4 (batch groups of 16 rows), block 1024
// (32 waves). Wave w owns gate col-tiles n0=w, n1=w+32. h kept in LDS in
// A-fragment bf16 layout; c kept in registers (4 elems/thread); gates via LDS.
// ---------------------------------------------------------------------------
__global__ __launch_bounds__(1024) void lstm_rec_kernel(
    const float* __restrict__ Xf, const unsigned short* __restrict__ whb,
    float* __restrict__ cs, float* __restrict__ hn) {
  __shared__ __align__(16) unsigned short h_swz[8][32][16];  // 8 KB
  __shared__ float gates[16][G4];                            // 64 KB
  int bg = blockIdx.x, tid = threadIdx.x, wave = tid >> 5, lane = tid & 31;

  ((unsigned long long*)h_swz)[tid] = 0ull;  // h0 = 0 (8192 B / 8 = 1024)
  float creg[4] = {0.f, 0.f, 0.f, 0.f};      // c0 = 0
  __syncthreads();

  const int n0 = wave, n1 = wave + 32;
  const int Nl = lane & 15, Mo = 8 * (lane >> 4);
  // Per-t stride in the fragment buffer: 64 tiles * 32 lanes * 8 floats.
  const size_t tstep = (size_t)64 * 32 * 8;
  const float* xp0 = Xf + (((size_t)bg * Tn * 64 + n0) * 32 + lane) * 8;
  const float* xp1 = Xf + (((size_t)bg * Tn * 64 + n1) * 32 + lane) * 8;

  for (int t = 0; t < Tn; ++t) {
    // Accumulator init = precomputed x-projection fragments (b128 loads).
    v8f acc0 = join8(*(const v4f*)xp0, *((const v4f*)xp0 + 1));
    v8f acc1 = join8(*(const v4f*)xp1, *((const v4f*)xp1 + 1));
    if (t + 1 < Tn) {  // pull next step's fragments toward the WGP
      __builtin_prefetch(xp0 + tstep, 0, 3);
      __builtin_prefetch(xp1 + tstep, 0, 3);
    }
    // gates += h @ W_h2h.T  (K = 256 in 8 steps of 32)
#pragma unroll
    for (int kk = 0; kk < 8; ++kk) {
      FragB a, b0, b1;
      a.q[0] = *(const v4u*)&h_swz[kk][lane][0];
      a.q[1] = *(const v4u*)&h_swz[kk][lane][8];
      const v4u* w0 = (const v4u*)(whb + (((n0 * 8 + kk) * 32 + lane) << 4));
      const v4u* w1 = (const v4u*)(whb + (((n1 * 8 + kk) * 32 + lane) << 4));
      b0.q[0] = w0[0]; b0.q[1] = w0[1];
      b1.q[0] = w1[0]; b1.q[1] = w1[1];
      acc0 = wmma_bf16(a, b0, acc0);
      acc1 = wmma_bf16(a, b1, acc1);
    }
#pragma unroll
    for (int r = 0; r < 8; ++r) {
      gates[Mo + r][16 * n0 + Nl] = acc0[r];
      gates[Mo + r][16 * n1 + Nl] = acc1[r];
    }
    __syncthreads();

    // Elementwise LSTM cell: 4096 elems, 4 per thread.
#pragma unroll
    for (int j = 0; j < 4; ++j) {
      int idx = j * 1024 + tid;
      int M = idx >> 8, col = idx & 255;
      float gi = gates[M][col];
      float gf = gates[M][col + 256];
      float gg = gates[M][col + 512];
      float go = gates[M][col + 768];
      float iv = sigf(gi), fv = sigf(gf), gv = tanhfast(gg), ov = sigf(go);
      float c = creg[j] * fv + iv * gv;
      creg[j] = c;
      cs[((size_t)(bg * 16 + M) * Tn + t) * Hn + col] = c;  // cn_outs
      float hv = ov * tanhfast(c);
      int l2, e;
      amap(M, col & 31, l2, e);
      h_swz[col >> 5][l2][e] = f2bf(hv);
      if (t == Tn - 1) hn[(bg * 16 + M) * Hn + col] = hv;
    }
    __syncthreads();
    xp0 += tstep;
    xp1 += tstep;
  }
}

// ---------------------------------------------------------------------------
// K3: out = hn @ W_fc.T + b_fc. grid 4, block 256 (8 waves = 8 col tiles).
// ---------------------------------------------------------------------------
__global__ __launch_bounds__(256) void fc_out_kernel(
    const float* __restrict__ hn, const unsigned short* __restrict__ wfb,
    const float* __restrict__ bfc, float* __restrict__ out) {
  __shared__ __align__(16) unsigned short a_swz[8][32][16];
  int bg = blockIdx.x, tid = threadIdx.x, wave = tid >> 5, lane = tid & 31;
  {
    int M = tid >> 4, c0 = (tid & 15) * 16;
    const float* hrow = hn + (bg * 16 + M) * Hn + c0;
#pragma unroll
    for (int j = 0; j < 16; ++j) {
      int col = c0 + j;
      int l2, e;
      amap(M, col & 31, l2, e);
      a_swz[col >> 5][l2][e] = f2bf(hrow[j]);
    }
  }
  __syncthreads();

  int n = wave;  // O = 128 -> 8 tiles of 16
  int Nl = lane & 15, Mo = 8 * (lane >> 4);
  float bb = bfc[16 * n + Nl];
  v8f acc;
#pragma unroll
  for (int r = 0; r < 8; ++r) acc[r] = bb;
#pragma unroll
  for (int kk = 0; kk < 8; ++kk) {
    FragB a, b;
    a.q[0] = *(const v4u*)&a_swz[kk][lane][0];
    a.q[1] = *(const v4u*)&a_swz[kk][lane][8];
    const v4u* wp = (const v4u*)(wfb + (((n * 8 + kk) * 32 + lane) << 4));
    b.q[0] = wp[0];
    b.q[1] = wp[1];
    acc = wmma_bf16(a, b, acc);
  }
#pragma unroll
  for (int r = 0; r < 8; ++r)
    out[(bg * 16 + Mo + r) * On + 16 * n + Nl] = acc[r];
}

// ---------------------------------------------------------------------------
extern "C" void kernel_launch(void* const* d_in, const int* in_sizes, int n_in,
                              void* d_out, int out_size, void* d_ws,
                              size_t ws_size, hipStream_t stream) {
  (void)in_sizes; (void)n_in; (void)out_size; (void)ws_size;
  const float* x     = (const float*)d_in[0];
  const float* W_x2h = (const float*)d_in[1];
  const float* b_x2h = (const float*)d_in[2];
  const float* W_h2h = (const float*)d_in[3];
  const float* b_h2h = (const float*)d_in[4];
  const float* W_fc  = (const float*)d_in[5];
  const float* b_fc  = (const float*)d_in[6];

  float* out = (float*)d_out;            // [64,128]
  float* cs  = out + (size_t)Bsz * On;   // [64,512,256]

  char* ws = (char*)d_ws;
  float* Xf = (float*)ws;                                       // 128 MB
  size_t off = (size_t)Bsz * Tn * G4 * sizeof(float);
  unsigned short* wxb = (unsigned short*)(ws + off); off += (size_t)64 * 4096 * 2;
  unsigned short* whb = (unsigned short*)(ws + off); off += (size_t)64 * 4096 * 2;
  unsigned short* wfb = (unsigned short*)(ws + off); off += (size_t)8 * 4096 * 2;
  float* hn = (float*)(ws + off);                               // [64,256]

  wfrag_convert<<<64 * 16, 256, 0, stream>>>(W_x2h, wxb, 64);
  wfrag_convert<<<64 * 16, 256, 0, stream>>>(W_h2h, whb, 64);
  wfrag_convert<<<8 * 16, 256, 0, stream>>>(W_fc, wfb, 8);
  xproj_kernel<<<dim3(Bsz * Tn / 16, 8), 256, 0, stream>>>(x, wxb, b_x2h,
                                                           b_h2h, Xf);
  lstm_rec_kernel<<<4, 1024, 0, stream>>>(Xf, whb, cs, hn);
  fc_out_kernel<<<4, 256, 0, stream>>>(hn, wfb, b_fc, out);
}
